// HyenaOperator_31293131719047
// MI455X (gfx1250) — matrix operand order repaired and checked by reading, not compiled, not run on hardware
//
#include <hip/hip_runtime.h>
#include <math.h>
#include <stdint.h>

// ---------------------------------------------------------------------------
// Hyena forward on MI455X (gfx1250, wave32, WMMA 16x16x32 f16->f32).
//
//  K0 k_cvt16      : one-time f32->f16 of u / w_in / w_out (cleans GEMM loop)
//  K1 k_in_proj    : up_t[b,e,l] = (u @ w_in^T + b_in)   WMMA f16, 51.5 GFLOP
//  K2 k_filter_mlp : H[l,64] = sin-MLP(z)                WMMA (64-wide layers)
//  K3 k_filter_proj: KF[c,l] = (H @ wf^T) * exp(-t|d_c|) WMMA
//  K4 k_conv       : fused dwconv3 gating + direct causal long conv (69 GFLOP,
//                    per-channel Toeplitz matvec -> f32 VALU sliding window).
//                    KF row staged to LDS by the Tensor Data Mover
//                    (tensor_load_to_lds + s_wait_tensorcnt).
//  K5 k_out_proj   : out = (G16 @ w_out^T + b_out)       WMMA f16, 17.2 GFLOP
// ---------------------------------------------------------------------------

typedef __attribute__((ext_vector_type(16))) _Float16 v16h;
typedef __attribute__((ext_vector_type(8)))  _Float16 v8h;
typedef __attribute__((ext_vector_type(8)))  float    v8f;
typedef unsigned int u32;
typedef __attribute__((ext_vector_type(4))) u32 u32x4;
typedef __attribute__((ext_vector_type(8))) u32 u32x8;

#define LSEQ 4096
#define DMODEL 1024
#define IW 3072

__device__ __forceinline__ v8f wmma32(v16h a, v16h b, v8f c) {
  return __builtin_amdgcn_wmma_f32_16x16x32_f16(false, a, false, b, (short)0, c,
                                                false, false);
}

// ---- f16-source fragment loaders (big GEMMs) -------------------------------
// A 16x32: lanes 0-15 -> M=lane, K 0-7 & 16-23 ; lanes 16-31 -> K 8-15 & 24-31.
__device__ __forceinline__ v16h load_a16(const _Float16* p0, int lda, int lane) {
  const _Float16* p = p0 + (size_t)(lane & 15) * lda + ((lane >> 4) << 3);
  v8h lo = *(const v8h*)p;
  v8h hi = *(const v8h*)(p + 16);
  v16h a;
#pragma unroll
  for (int j = 0; j < 8; ++j) { a[j] = lo[j]; a[8 + j] = hi[j]; }
  return a;
}
// B 32x16 from N-major [n][k]: lanes 0-15 -> N=lane, K 0-7 & 16-23 ; 16-31 -> K 8-15 & 24-31.
__device__ __forceinline__ v16h load_b16(const _Float16* p0, int ldb, int lane) {
  const _Float16* p = p0 + (size_t)(lane & 15) * ldb + ((lane >> 4) << 3);
  v8h lo = *(const v8h*)p;
  v8h hi = *(const v8h*)(p + 16);
  v16h b;
#pragma unroll
  for (int j = 0; j < 8; ++j) { b[j] = lo[j]; b[8 + j] = hi[j]; }
  return b;
}
// ---- f32-source fragment loaders (small filter kernels) --------------------
__device__ __forceinline__ v16h load_a32(const float* p0, int lda, int lane) {
  const float* p = p0 + (size_t)(lane & 15) * lda + ((lane >> 4) << 3);
  v16h a;
#pragma unroll
  for (int j = 0; j < 8; ++j) a[j] = (_Float16)p[j];
#pragma unroll
  for (int j = 0; j < 8; ++j) a[8 + j] = (_Float16)p[16 + j];
  return a;
}
__device__ __forceinline__ v16h load_b32(const float* p0, int ldb, int lane) {
  const float* p = p0 + (size_t)(lane & 15) * ldb + ((lane >> 4) << 3);
  v16h b;
#pragma unroll
  for (int j = 0; j < 16; ++j) b[j] = (_Float16)p[j < 8 ? j : j + 8];
  return b;
}

// -------------------- K0: bulk f32 -> f16 conversion ------------------------
__global__ __launch_bounds__(256) void k_cvt16(const float* __restrict__ s,
                                               _Float16* __restrict__ d) {
  const int i = (blockIdx.x * 256 + threadIdx.x) * 4;
  const float4 v = *(const float4*)(s + i);
  d[i + 0] = (_Float16)v.x;
  d[i + 1] = (_Float16)v.y;
  d[i + 2] = (_Float16)v.z;
  d[i + 3] = (_Float16)v.w;
}

// -------------------- K1: input projection (store transposed) ---------------
__global__ __launch_bounds__(256) void k_in_proj(
    const _Float16* __restrict__ U16, const _Float16* __restrict__ Win16,
    const float* __restrict__ Bin, float* __restrict__ upT) {
  const int lane = threadIdx.x & 31, wid = threadIdx.x >> 5;
  const int m0 = blockIdx.x * 64 + (wid >> 2) * 32;   // rows [0,8192)
  const int n0 = blockIdx.y * 256 + (wid & 3) * 64;   // cols [0,3072)
  v8f acc[2][4] = {};
  for (int k0 = 0; k0 < DMODEL; k0 += 32) {
    if (k0 + 32 < DMODEL)
      __builtin_prefetch(U16 + (size_t)m0 * DMODEL + k0 + 32, 0, 1);
    v16h a0 = load_a16(U16 + (size_t)m0 * DMODEL + k0, DMODEL, lane);
    v16h a1 = load_a16(U16 + (size_t)(m0 + 16) * DMODEL + k0, DMODEL, lane);
#pragma unroll
    for (int t = 0; t < 4; ++t) {
      v16h b = load_b16(Win16 + (size_t)(n0 + t * 16) * DMODEL + k0, DMODEL, lane);
      acc[0][t] = wmma32(a0, b, acc[0][t]);
      acc[1][t] = wmma32(a1, b, acc[1][t]);
    }
  }
  const int hi = lane >> 4, nn = lane & 15;
#pragma unroll
  for (int mi = 0; mi < 2; ++mi) {
#pragma unroll
    for (int t = 0; t < 4; ++t) {
      const int e = n0 + t * 16 + nn;
      const float bias = Bin[e];
#pragma unroll
      for (int r = 0; r < 8; ++r) {
        const int gm = m0 + mi * 16 + r + 8 * hi;
        const int b = gm >> 12, l = gm & (LSEQ - 1);
        upT[((size_t)b * IW + e) * LSEQ + l] = acc[mi][t][r] + bias;
      }
    }
  }
}

// -------------------- K2: implicit-filter MLP (sin) --------------------------
__global__ __launch_bounds__(256) void k_filter_mlp(
    const float* __restrict__ Z, const float* __restrict__ W0,
    const float* __restrict__ B0, const float* __restrict__ FRQ,
    const float* __restrict__ W1, const float* __restrict__ B1,
    const float* __restrict__ W2, const float* __restrict__ B2,
    float* __restrict__ H) {
  __shared__ float h[128 * 72];
  const int l0g = blockIdx.x * 128;
  const int lane = threadIdx.x & 31, wid = threadIdx.x >> 5;
  const int m0 = wid * 16;
  const int hi = lane >> 4, nn = lane & 15;

  for (int idx = threadIdx.x; idx < 128 * 64; idx += 256) {
    const int ll = idx >> 6, j = idx & 63;
    const int l = l0g + ll;
    const float pre = Z[l * 3 + 0] * W0[j * 3 + 0] +
                      Z[l * 3 + 1] * W0[j * 3 + 1] +
                      Z[l * 3 + 2] * W0[j * 3 + 2] + B0[j];
    h[ll * 72 + j] = sinf(FRQ[j] * pre);
  }
  __syncthreads();

  {  // layer 1: in place through LDS
    v16h a0 = load_a32(h + m0 * 72 + 0, 72, lane);
    v16h a1 = load_a32(h + m0 * 72 + 32, 72, lane);
    v8f acc[4] = {};
#pragma unroll
    for (int t = 0; t < 4; ++t) {
      acc[t] = wmma32(a0, load_b32(W1 + (size_t)(t * 16) * 64 + 0, 64, lane), acc[t]);
      acc[t] = wmma32(a1, load_b32(W1 + (size_t)(t * 16) * 64 + 32, 64, lane), acc[t]);
    }
    __syncthreads();
#pragma unroll
    for (int t = 0; t < 4; ++t) {
      const int n = t * 16 + nn;
      const float fq = FRQ[n], bb = B1[n];
#pragma unroll
      for (int rr = 0; rr < 8; ++rr)
        h[(m0 + rr + 8 * hi) * 72 + n] = sinf(fq * (acc[t][rr] + bb));
    }
  }
  __syncthreads();

  {  // layer 2 -> global H
    v16h a0 = load_a32(h + m0 * 72 + 0, 72, lane);
    v16h a1 = load_a32(h + m0 * 72 + 32, 72, lane);
    v8f acc[4] = {};
#pragma unroll
    for (int t = 0; t < 4; ++t) {
      acc[t] = wmma32(a0, load_b32(W2 + (size_t)(t * 16) * 64 + 0, 64, lane), acc[t]);
      acc[t] = wmma32(a1, load_b32(W2 + (size_t)(t * 16) * 64 + 32, 64, lane), acc[t]);
    }
#pragma unroll
    for (int t = 0; t < 4; ++t) {
      const int n = t * 16 + nn;
      const float fq = FRQ[n], bb = B2[n];
#pragma unroll
      for (int rr = 0; rr < 8; ++rr) {
        const int m = m0 + rr + 8 * hi;
        H[(size_t)(l0g + m) * 64 + n] = sinf(fq * (acc[t][rr] + bb));
      }
    }
  }
}

// -------------------- K3: filter projection + exp modulation ----------------
__global__ __launch_bounds__(256) void k_filter_proj(
    const float* __restrict__ H, const float* __restrict__ WF,
    float* __restrict__ KF) {
  const int lane = threadIdx.x & 31, wid = threadIdx.x >> 5;
  const int m0 = blockIdx.x * 32 + (wid >> 2) * 16;   // l
  const int n0 = blockIdx.y * 256 + (wid & 3) * 64;   // channel
  v8f acc[4] = {};
#pragma unroll
  for (int k0 = 0; k0 < 64; k0 += 32) {
    v16h a = load_a32(H + (size_t)m0 * 64 + k0, 64, lane);
#pragma unroll
    for (int t = 0; t < 4; ++t)
      acc[t] = wmma32(a, load_b32(WF + (size_t)(n0 + t * 16) * 64 + k0, 64, lane),
                      acc[t]);
  }
  const int hi = lane >> 4, nn = lane & 15;
  const float mind = -3.0701134573253945f;   // ln(0.01)/1.5
  const float maxd = -15.350567286626973f;   // ln(0.01)/0.3
#pragma unroll
  for (int t = 0; t < 4; ++t) {
    const int e = n0 + t * 16 + nn;
    const float delta = fabsf(mind + (maxd - mind) * ((float)e * (1.0f / 1023.0f)));
#pragma unroll
    for (int rr = 0; rr < 8; ++rr) {
      const int l = m0 + rr + 8 * hi;
      const float tt = (float)l * (1.0f / 4095.0f);
      KF[(size_t)e * LSEQ + l] = acc[t][rr] * expf(-tt * delta);  // [c,l]
    }
  }
}

// -------------------- K4: fused gating + direct causal long conv ------------
__global__ __launch_bounds__(256) void k_conv(
    const float* __restrict__ upT, const float* __restrict__ KF,
    const float* __restrict__ SFW, const float* __restrict__ SFB,
    const float* __restrict__ Dv, _Float16* __restrict__ G16) {
  __shared__ float lx[2 * LSEQ];   // [0,L): zero pad, [L,2L): vx1 = v*x1
  __shared__ float lk[LSEQ];
  const int b = blockIdx.x >> 10;
  const int c = blockIdx.x & 1023;
  const int tid = threadIdx.x;

  // ---- TDM: DMA the 16KB filter row KF[c,:] into LDS (wave 0 issues) ------
  if (tid < 32) {
    const u32 lds_addr = (u32)(uintptr_t)(&lk[0]);
    const unsigned long long ga =
        (unsigned long long)(uintptr_t)(KF + (size_t)c * LSEQ);
    u32x4 g0;
    g0[0] = 1u;                                            // count=1 (valid D#)
    g0[1] = lds_addr;                                      // lds_addr
    g0[2] = (u32)(ga & 0xFFFFFFFFu);                       // global_addr[31:0]
    g0[3] = (u32)((ga >> 32) & 0x01FFFFFFu) | (2u << 30);  // addr[56:32]|type=2
    u32x8 g1;
    g1[0] = (2u << 16);          // workgroup_mask=0, data_size=4B
    g1[1] = (LSEQ & 0xFFFFu) << 16;   // tensor_dim0[15:0] @ bits 63:48
    g1[2] = (LSEQ >> 16) | (1u << 16);// tensor_dim0[31:16] | tensor_dim1=1
    g1[3] = (u32)LSEQ << 16;     // tile_dim0 = 4096 @ bits 127:112
    g1[4] = 1u;                  // tile_dim1 = 1
    g1[5] = (u32)LSEQ;           // tensor_dim0_stride[31:0]
    g1[6] = 0u;
    g1[7] = 0u;
    asm volatile("tensor_load_to_lds %0, %1" : : "s"(g0), "s"(g1) : "memory");
    __builtin_amdgcn_s_wait_tensorcnt(0);
  }

  const float* r1 = upT + ((size_t)b * IW + DMODEL + c) * LSEQ;       // x1 src
  const float* r2 = upT + ((size_t)b * IW + 2 * DMODEL + c) * LSEQ;   // v  src
  const int e1 = DMODEL + c, e2 = 2 * DMODEL + c;
  const float a10 = SFW[e1 * 3], a11 = SFW[e1 * 3 + 1], a12 = SFW[e1 * 3 + 2];
  const float a20 = SFW[e2 * 3], a21 = SFW[e2 * 3 + 1], a22 = SFW[e2 * 3 + 2];
  const float c1 = SFB[e1], c2 = SFB[e2];

  for (int l = tid; l < LSEQ; l += 256) {
    float x1v = c1 + a12 * r1[l];
    float vv  = c2 + a22 * r2[l];
    if (l >= 1) { x1v += a11 * r1[l - 1]; vv += a21 * r2[l - 1]; }
    if (l >= 2) { x1v += a10 * r1[l - 2]; vv += a20 * r2[l - 2]; }
    lx[l] = 0.0f;
    lx[LSEQ + l] = x1v * vv;
  }
  __syncthreads();

  // y[l] = sum_{d} k[d] * xpad[L + l - d]; 16 outputs/lane, sliding window.
  const int l0 = tid * 16;
  float acc[16];
#pragma unroll
  for (int i = 0; i < 16; ++i) acc[i] = 0.0f;
  float r[16];                     // circular window: r[idx & 15] = xpad[idx]
#pragma unroll
  for (int j = 0; j < 16; ++j) r[j] = lx[LSEQ + l0 + j];

  for (int dblk = 0; dblk < LSEQ; dblk += 16) {
#pragma unroll
    for (int s = 0; s < 16; ++s) {
      const float kv = lk[dblk + s];
#pragma unroll
      for (int i = 0; i < 16; ++i)
        acc[i] = fmaf(kv, r[(i - s) & 15], acc[i]);
      r[15 - s] = lx[LSEQ + l0 - (dblk + s) - 1];   // slide window down by 1
    }
  }

  const float* r0 = upT + ((size_t)b * IW + c) * LSEQ;                // x0 src
  const float a00 = SFW[c * 3], a01 = SFW[c * 3 + 1], a02 = SFW[c * 3 + 2];
  const float c0 = SFB[c], Dc = Dv[c];
#pragma unroll
  for (int i = 0; i < 16; ++i) {
    const int l = l0 + i;
    float x0v = c0 + a02 * r0[l];
    if (l >= 1) x0v += a01 * r0[l - 1];
    if (l >= 2) x0v += a00 * r0[l - 2];
    const float y = acc[i] + lx[LSEQ + l] * Dc;
    G16[((size_t)b * LSEQ + l) * DMODEL + c] = (_Float16)(y * x0v);
  }
}

// -------------------- K5: output projection ---------------------------------
__global__ __launch_bounds__(256) void k_out_proj(
    const _Float16* __restrict__ G16, const _Float16* __restrict__ Wout16,
    const float* __restrict__ Bout, float* __restrict__ Out) {
  const int lane = threadIdx.x & 31, wid = threadIdx.x >> 5;
  const int m0 = blockIdx.x * 64 + (wid >> 2) * 32;
  const int n0 = blockIdx.y * 256 + (wid & 3) * 64;
  v8f acc[2][4] = {};
  for (int k0 = 0; k0 < DMODEL; k0 += 32) {
    if (k0 + 32 < DMODEL)
      __builtin_prefetch(G16 + (size_t)m0 * DMODEL + k0 + 32, 0, 1);
    v16h a0 = load_a16(G16 + (size_t)m0 * DMODEL + k0, DMODEL, lane);
    v16h a1 = load_a16(G16 + (size_t)(m0 + 16) * DMODEL + k0, DMODEL, lane);
#pragma unroll
    for (int t = 0; t < 4; ++t) {
      v16h b = load_b16(Wout16 + (size_t)(n0 + t * 16) * DMODEL + k0, DMODEL, lane);
      acc[0][t] = wmma32(a0, b, acc[0][t]);
      acc[1][t] = wmma32(a1, b, acc[1][t]);
    }
  }
  const int hi = lane >> 4, nn = lane & 15;
#pragma unroll
  for (int mi = 0; mi < 2; ++mi) {
#pragma unroll
    for (int t = 0; t < 4; ++t) {
      const int e = n0 + t * 16 + nn;
      const float bias = Bout[e];
#pragma unroll
      for (int r = 0; r < 8; ++r) {
        const int gm = m0 + mi * 16 + r + 8 * hi;
        Out[(size_t)gm * DMODEL + e] = acc[mi][t][r] + bias;
      }
    }
  }
}

// ---------------------------------------------------------------------------
extern "C" void kernel_launch(void* const* d_in, const int* in_sizes, int n_in,
                              void* d_out, int out_size, void* d_ws,
                              size_t ws_size, hipStream_t stream) {
  (void)in_sizes; (void)n_in; (void)out_size; (void)ws_size;
  const float* u     = (const float*)d_in[0];
  const float* w_in  = (const float*)d_in[1];
  const float* b_in  = (const float*)d_in[2];
  const float* sf_w  = (const float*)d_in[3];
  const float* sf_b  = (const float*)d_in[4];
  const float* z     = (const float*)d_in[5];
  const float* w0    = (const float*)d_in[6];
  const float* b0    = (const float*)d_in[7];
  const float* freq  = (const float*)d_in[8];
  const float* w1    = (const float*)d_in[9];
  const float* b1    = (const float*)d_in[10];
  const float* w2    = (const float*)d_in[11];
  const float* b2    = (const float*)d_in[12];
  const float* wf    = (const float*)d_in[13];
  const float* Dv    = (const float*)d_in[14];
  const float* w_out = (const float*)d_in[15];
  const float* b_out = (const float*)d_in[16];
  float* out = (float*)d_out;

  char* wsb = (char*)d_ws;
  float*    upT  = (float*)(wsb);                    // 100,663,296 B
  float*    KF   = (float*)(wsb + 100663296);        //  16,777,216 B
  float*    H    = (float*)(wsb + 117440512);        //   1,048,576 B
  _Float16* u16  = (_Float16*)(wsb + 118489088);     //  16,777,216 B
  _Float16* wi16 = (_Float16*)(wsb + 135266304);     //   6,291,456 B
  _Float16* wo16 = (_Float16*)(wsb + 141557760);     //   2,097,152 B
  _Float16* G16  = (_Float16*)(wsb + 143654912);     //  16,777,216 B

  dim3 blk(256);
  k_cvt16      <<<dim3(8192),    blk, 0, stream>>>(u, u16);
  k_cvt16      <<<dim3(3072),    blk, 0, stream>>>(w_in, wi16);
  k_cvt16      <<<dim3(1024),    blk, 0, stream>>>(w_out, wo16);
  k_in_proj    <<<dim3(128, 12), blk, 0, stream>>>(u16, wi16, b_in, upT);
  k_filter_mlp <<<dim3(32),      blk, 0, stream>>>(z, w0, b0, freq, w1, b1, w2, b2, H);
  k_filter_proj<<<dim3(128, 4),  blk, 0, stream>>>(H, wf, KF);
  k_conv       <<<dim3(2048),    blk, 0, stream>>>(upT, KF, sf_w, sf_b, Dv, G16);
  k_out_proj   <<<dim3(128, 4),  blk, 0, stream>>>(G16, wo16, b_out, out);
}
